// CAM_Module_72868415143945
// MI455X (gfx1250) — matrix operand order repaired and checked by reading, not compiled, not run on hardware
//
#include <hip/hip_runtime.h>

// ---------------------------------------------------------------------------
// CAM module on MI455X (gfx1250, wave32, WMMA), LDS-staged block GEMMs.
//   1. cvt q,k  : fp32 -> bf16                          (ws: qb, kb)
//   2. cvt+T v  : fp32 -> bf16 transposed pv^T[n,d]     (ws: vT)
//   3. GEMM1    : E[kc] = partial (qb . kb^T), split-K x4, 128x128 blocks
//   4. softmax  : attn = exp(min-e)/sum over sum of 4 partials -> bf16
//   5. GEMM2    : out = gamma * (At . pv) + v, 128x128 blocks
// Workspace: qb(32MB) kb(32MB) vT(32MB) E(4x8MB) At(4MB) = 132MB.
// ---------------------------------------------------------------------------

typedef __attribute__((ext_vector_type(16))) __bf16 v16bf;
typedef __attribute__((ext_vector_type(8)))  __bf16 v8bf;
typedef __attribute__((ext_vector_type(4)))  __bf16 v4bf;
typedef __attribute__((ext_vector_type(8)))  float  v8f;

static constexpr int Bn = 8;     // batches
static constexpr int Cc = 512;   // channels
static constexpr int Nn = 4096;  // H*W
static constexpr int LDSP = 40;  // LDS row stride (elems): 80B -> conflict-free frags

// ---- fp32 -> bf16 elementwise (x4 vectorized) -----------------------------
__global__ __launch_bounds__(256) void cvt_bf16_x4(const float* __restrict__ in,
                                                   __bf16* __restrict__ out) {
    size_t i = ((size_t)blockIdx.x * 256 + threadIdx.x) * 4;
    float4 f = *(const float4*)(in + i);
    v4bf o = {(__bf16)f.x, (__bf16)f.y, (__bf16)f.z, (__bf16)f.w};
    *(v4bf*)(out + i) = o;
}

// ---- fp32 v[b,c,n] -> bf16 vT[b,n,c] (32x32 LDS tiles) --------------------
__global__ __launch_bounds__(256) void cvt_transpose_v(const float* __restrict__ V,
                                                       __bf16* __restrict__ VT) {
    __shared__ __bf16 tile[32][33];
    const float* Vb = V + (size_t)blockIdx.z * Cc * Nn;
    __bf16* VTb = VT + (size_t)blockIdx.z * Nn * Cc;
    int n0 = blockIdx.x * 32, c0 = blockIdx.y * 32;
    int tx = threadIdx.x & 31, ty = threadIdx.x >> 5;   // 32 x 8
#pragma unroll
    for (int j = 0; j < 32; j += 8)
        tile[ty + j][tx] = (__bf16)Vb[(size_t)(c0 + ty + j) * Nn + n0 + tx];
    __syncthreads();
#pragma unroll
    for (int j = 0; j < 32; j += 8)
        VTb[(size_t)(n0 + ty + j) * Cc + c0 + tx] = tile[tx][ty + j];
}

__device__ inline v16bf cat8(v8bf lo, v8bf hi) {
    return __builtin_shufflevector(lo, hi, 0, 1, 2, 3, 4, 5, 6, 7,
                                   8, 9, 10, 11, 12, 13, 14, 15);
}

// ---------------------------------------------------------------------------
// Shared GEMM core: 256 threads (8 waves, 2x4), block tile 128(M)x128(N),
// K-step 32, both operands staged through LDS. A and B^T are row-major with
// K contiguous (leading dims lda/ldb); pointers pre-offset to the block row.
// Wave w: rows [ (w>>2)*64, +64 ), cols [ (w&3)*32, +32 ).
// A-frag per ISA 16-bit A layout: lane r rows, half-wave g K-octets {8g,16+8g}.
// B-frag: lane holds col n, same K-octets as A (fed from transposed-B rows).
// ---------------------------------------------------------------------------
__device__ inline void gemm_core_128x128(const __bf16* __restrict__ Ag,
                                         const __bf16* __restrict__ Bg,
                                         int lda, int ldb, int kBeg, int kEnd,
                                         __bf16* As, __bf16* Bs, v8f acc[4][2]) {
    const int tid = threadIdx.x;
    const int lane = tid & 31, wave = tid >> 5;
    const int g = lane >> 4, r = lane & 15;
    const int wm = (wave >> 2) * 64, wn = (wave & 3) * 32;
    const int lrow = tid >> 1;            // 0..127: staging row
    const int lhalf = (tid & 1) * 16;     // 0 or 16: staging K-half

    for (int k = kBeg; k < kEnd; k += 32) {
        // global -> regs (2x b128 each for A and B tile slices)
        const __bf16* га = Ag + (size_t)lrow * lda + k + lhalf;
        const __bf16* гб = Bg + (size_t)lrow * ldb + k + lhalf;
        v8bf a0 = *(const v8bf*)(га), a1 = *(const v8bf*)(га + 8);
        v8bf b0 = *(const v8bf*)(гб), b1 = *(const v8bf*)(гб + 8);
        __syncthreads();                  // previous iter's LDS reads done
        *(v8bf*)(As + lrow * LDSP + lhalf) = a0;
        *(v8bf*)(As + lrow * LDSP + lhalf + 8) = a1;
        *(v8bf*)(Bs + lrow * LDSP + lhalf) = b0;
        *(v8bf*)(Bs + lrow * LDSP + lhalf + 8) = b1;
        __syncthreads();
        // LDS -> fragments
        v16bf af[4], bf[2];
#pragma unroll
        for (int i = 0; i < 4; ++i) {
            const __bf16* row = As + (wm + i * 16 + r) * LDSP;
            af[i] = cat8(*(const v8bf*)(row + g * 8),
                         *(const v8bf*)(row + 16 + g * 8));
        }
#pragma unroll
        for (int j = 0; j < 2; ++j) {
            const __bf16* row = Bs + (wn + j * 16 + r) * LDSP;
            bf[j] = cat8(*(const v8bf*)(row + g * 8),
                         *(const v8bf*)(row + 16 + g * 8));
        }
#pragma unroll
        for (int i = 0; i < 4; ++i)
#pragma unroll
            for (int j = 0; j < 2; ++j)
                acc[i][j] = __builtin_amdgcn_wmma_f32_16x16x32_bf16(
                    false, af[i], false, bf[j], (short)0, acc[i][j], false, false);
    }
}

// ---- GEMM1 (split-K x4): E[kc,b,c,d] = sum_{n in chunk} Q[b,c,n]*K[b,d,n] --
__global__ __launch_bounds__(256) void gemm1_qk(const __bf16* __restrict__ Q,
                                                const __bf16* __restrict__ K,
                                                float* __restrict__ E) {
    __shared__ __bf16 As[128 * LDSP], Bs[128 * LDSP];
    const int b = blockIdx.z >> 2, kc = blockIdx.z & 3;
    const __bf16* Ag = Q + (size_t)b * Cc * Nn + (size_t)blockIdx.y * 128 * Nn;
    const __bf16* Bg = K + (size_t)b * Cc * Nn + (size_t)blockIdx.x * 128 * Nn;
    v8f acc[4][2] = {};
    gemm_core_128x128(Ag, Bg, Nn, Nn, kc * 1024, kc * 1024 + 1024, As, Bs, acc);

    const int lane = threadIdx.x & 31, wave = threadIdx.x >> 5;
    const int g = lane >> 4, r = lane & 15;
    const int row0 = blockIdx.y * 128 + (wave >> 2) * 64;
    const int col0 = blockIdx.x * 128 + (wave & 3) * 32;
    float* Eb = E + ((size_t)kc * Bn + b) * Cc * Cc;
#pragma unroll
    for (int i = 0; i < 4; ++i)
#pragma unroll
        for (int j = 0; j < 2; ++j)
#pragma unroll
            for (int t = 0; t < 8; ++t)
                Eb[(size_t)(row0 + i * 16 + t + 8 * g) * Cc + col0 + j * 16 + r] =
                    acc[i][j][t];
}

// ---- softmax over rows: attn_d = exp(min_e - e_d)/sum, e = sum 4 partials --
__global__ __launch_bounds__(256) void softmax_rows(const float* __restrict__ E,
                                                    __bf16* __restrict__ A) {
    __shared__ float red[256];
    const size_t S = (size_t)Bn * Cc * Cc;      // partial stride
    const float* e = E + (size_t)blockIdx.x * Cc;
    __bf16* a = A + (size_t)blockIdx.x * Cc;
    int t = threadIdx.x;
    float e0 = e[t] + e[S + t] + e[2 * S + t] + e[3 * S + t];
    int t2 = t + 256;
    float e1 = e[t2] + e[S + t2] + e[2 * S + t2] + e[3 * S + t2];
    red[t] = fminf(e0, e1);
    __syncthreads();
#pragma unroll
    for (int s = 128; s > 0; s >>= 1) {
        if (t < s) red[t] = fminf(red[t], red[t + s]);
        __syncthreads();
    }
    float m = red[0];
    __syncthreads();
    float p0 = __expf(m - e0), p1 = __expf(m - e1);
    red[t] = p0 + p1;
    __syncthreads();
#pragma unroll
    for (int s = 128; s > 0; s >>= 1) {
        if (t < s) red[t] += red[t + s];
        __syncthreads();
    }
    float inv = 1.0f / red[0];
    a[t] = (__bf16)(p0 * inv);
    a[t2] = (__bf16)(p1 * inv);
}

// ---- GEMM2: out[b,c,n] = gamma * sum_d At[b,c,d]*pv[b,d,n] + v[b,c,n] -----
__global__ __launch_bounds__(256) void gemm2_av(const __bf16* __restrict__ At,
                                                const __bf16* __restrict__ VT,
                                                const float* __restrict__ Vorig,
                                                const float* __restrict__ gamma,
                                                float* __restrict__ Out) {
    __shared__ __bf16 As[128 * LDSP], Bs[128 * LDSP];
    const int b = blockIdx.z;
    const __bf16* Ag = At + (size_t)b * Cc * Cc + (size_t)blockIdx.y * 128 * Cc;
    const __bf16* Bg = VT + (size_t)b * Nn * Cc + (size_t)blockIdx.x * 128 * Cc;
    v8f acc[4][2] = {};
    gemm_core_128x128(Ag, Bg, Cc, Cc, 0, Cc, As, Bs, acc);

    const int lane = threadIdx.x & 31, wave = threadIdx.x >> 5;
    const int g = lane >> 4, r = lane & 15;
    const int row0 = blockIdx.y * 128 + (wave >> 2) * 64;   // channel rows
    const int col0 = blockIdx.x * 128 + (wave & 3) * 32;    // spatial cols
    const float gm = gamma[0];
    const float* Vb = Vorig + (size_t)b * Cc * Nn;
    float* Ob = Out + (size_t)b * Cc * Nn;
#pragma unroll
    for (int i = 0; i < 4; ++i)
#pragma unroll
        for (int j = 0; j < 2; ++j)
#pragma unroll
            for (int t = 0; t < 8; ++t) {
                size_t idx = (size_t)(row0 + i * 16 + t + 8 * g) * Nn + col0 + j * 16 + r;
                Ob[idx] = gm * acc[i][j][t] + Vb[idx];
            }
}

extern "C" void kernel_launch(void* const* d_in, const int* in_sizes, int n_in,
                              void* d_out, int out_size, void* d_ws, size_t ws_size,
                              hipStream_t stream) {
    const float* q = (const float*)d_in[0];
    const float* k = (const float*)d_in[1];
    const float* v = (const float*)d_in[2];
    const float* gamma = (const float*)d_in[3];
    float* out = (float*)d_out;

    // Workspace layout (needs ~132 MB)
    size_t qkv = (size_t)Bn * Cc * Nn;                  // 16.7M elems
    __bf16* qb = (__bf16*)d_ws;
    __bf16* kb = qb + qkv;
    __bf16* vT = kb + qkv;
    float*  E  = (float*)(vT + qkv);                    // 4 x B*C*C fp32 partials
    __bf16* At = (__bf16*)(E + 4 * (size_t)Bn * Cc * Cc);

    int cvt_blocks = (int)(qkv / 4 / 256);              // exact: 16384
    cvt_bf16_x4<<<cvt_blocks, 256, 0, stream>>>(q, qb);
    cvt_bf16_x4<<<cvt_blocks, 256, 0, stream>>>(k, kb);
    cvt_transpose_v<<<dim3(Nn / 32, Cc / 32, Bn), 256, 0, stream>>>(v, vT);

    gemm1_qk<<<dim3(Cc / 128, Cc / 128, Bn * 4), 256, 0, stream>>>(qb, kb, E);
    softmax_rows<<<Bn * Cc, 256, 0, stream>>>(E, At);
    gemm2_av<<<dim3(Nn / 128, Cc / 128, Bn), 256, 0, stream>>>(At, vT, v, gamma, out);
}
